// HeunIntegrator_74440373174357
// MI455X (gfx1250) — matrix operand order repaired and checked by reading, not compiled, not run on hardware
//
#include <hip/hip_runtime.h>
#include <math.h>

// ---------------------------------------------------------------------------
// HeunIntegrator on MI455X (gfx1250): bf16 WMMA + TDM-staged operands.
//   B=8192, D=1024, R=256, steps=16, dt=0.01
//   christoffel(v) = U @ (W v)^2 ; dv = force - christoffel(v)
//   x' = wrap(x + dt/2 * sum_steps(v_k + v_pred_k))   (wrap composes mod 2pi)
// ---------------------------------------------------------------------------

#define BDIM 8192
#define DDIM 1024
#define RDIM 256
#define DT   0.01f

typedef float  v8f     __attribute__((ext_vector_type(8)));
typedef __bf16 bf16x16 __attribute__((ext_vector_type(16)));
typedef __bf16 bf16x8  __attribute__((ext_vector_type(8)));
typedef unsigned int u32x4 __attribute__((ext_vector_type(4)));
typedef int          i32x8 __attribute__((ext_vector_type(8)));
typedef int          i32x4 __attribute__((ext_vector_type(4)));

union ABf { bf16x16 v; bf16x8 h[2]; };

// Padded W panel row pitch: 32 bf16 data + 8 bf16 TDM pad = 80 B
#define WP_PITCH 40

__device__ __forceinline__ unsigned short f2bf(float f) {
  unsigned u = __float_as_uint(f);
  u += 0x7FFFu + ((u >> 16) & 1u);
  return (unsigned short)(u >> 16);
}

__device__ __forceinline__ bf16x8 ld8(const unsigned short* p) {
  return *(const bf16x8*)p;
}

// ---------------------------------------------------------------------------
// TDM: DMA one 256-row x 32-col bf16 K-panel of W into LDS with 16B row pad.
// D# per CDNA5 ISA ch.8 (2-D tensor, groups 2/3 zero).
// ---------------------------------------------------------------------------
__device__ __forceinline__ void tdm_load_wpanel(unsigned short* ldsDst,
                                                const unsigned short* gsrc) {
  unsigned lds_addr = (unsigned)(unsigned long long)
      (__attribute__((address_space(3))) unsigned short*)ldsDst;
  unsigned long long ga = (unsigned long long)gsrc;

  u32x4 g0 = { 1u,                                   // count=1, user mode
               lds_addr,                              // lds_addr [63:32]
               (unsigned)ga,                          // global_addr lo
               ((unsigned)(ga >> 32) & 0x01FFFFFFu)   // global_addr hi (57b)
                 | 0x80000000u };                     // type=2 ("image")

  i32x8 g1 = { (int)((1u << 16)                      // data_size = 2 bytes
                   | (1u << 20)                      // pad_enable
                   | (3u << 22)                      // pad_interval: 16 DW = 64B
                   | (3u << 25)),                    // pad_amount:    4 DW = 16B
               (int)((DDIM & 0xFFFFu) << 16),        // tensor_dim0 lo16 @ [31:16]
               (int)((DDIM >> 16) | ((RDIM & 0xFFFFu) << 16)), // td0 hi | td1 lo
               (int)((RDIM >> 16) | (32u << 16)),    // td1 hi | tile_dim0 = 32
               (int)RDIM,                            // tile_dim1=256, tile_dim2=0
               (int)DDIM,                            // tensor_dim0_stride lo32
               0, 0 };
  i32x4 z4 = { 0, 0, 0, 0 };
  i32x8 z8 = { 0, 0, 0, 0, 0, 0, 0, 0 };
  __builtin_amdgcn_tensor_load_to_lds(g0, g1, z4, z4, z8, 0);
}

// ---------------------------------------------------------------------------
// Fused Christoffel + Heun half-step. Block = 64 threads = 2 waves,
// each wave owns one 16-row M-tile; W panels TDM-staged & shared in LDS.
// ---------------------------------------------------------------------------
template<int PHASE>
__global__ __launch_bounds__(64)
void christoffel_step(const unsigned short* __restrict__ vin_bf,   // [B][D] bf16
                      float*                __restrict__ vmaster,  // [B][D] f32 v
                      const float*          __restrict__ force,    // [B][D]
                      const unsigned short* __restrict__ Wbf,      // [R][D] bf16
                      const unsigned short* __restrict__ Ubf,      // [D][R] bf16
                      float*                __restrict__ dv1buf,   // [B][D]
                      float*                __restrict__ Sacc,     // [B][D]
                      unsigned short*       __restrict__ vout_bf)  // [B][D] bf16
{
  __shared__ unsigned short wpanel[2][RDIM][WP_PITCH]; // 2 x 20KB, TDM-padded
  __shared__ unsigned short qlds[2][16][RDIM];         // per-wave (Wv)^2, 16KB

  const int lane = threadIdx.x & 31;
  const int wv   = threadIdx.x >> 5;
  const int hlf  = lane >> 4;
  const int l15  = lane & 15;
  const int row0 = (blockIdx.x * 2 + wv) * 16;

  // Prologue: DMA first W K-panel (wave 0 drives the TDM for the block).
  if (wv == 0) {
    tdm_load_wpanel(&wpanel[0][0][0], Wbf);
    __builtin_amdgcn_s_wait_tensorcnt(0);
  }
  __syncthreads();

  // ---------------- GEMM1: p[16 x 256] = v_tile @ W^T ------------------------
  v8f zero = {};
  v8f accP[16];
#pragma unroll
  for (int nt = 0; nt < 16; ++nt) accP[nt] = zero;

  // A fragment (16-bit A 16x32): lane l15 = row; hlf picks K 0-15/16-31
  const unsigned short* aRow = vin_bf + (size_t)(row0 + l15) * DDIM + hlf * 16;
  ABf aBuf[2];
  aBuf[0].h[0] = ld8(aRow);
  aBuf[0].h[1] = ld8(aRow + 8);

  for (int c = 0; c < 32; ++c) {
    const int buf = c & 1;
    // Kick off DMA of next K-panel; overlaps with this chunk's 16 WMMAs.
    if ((c + 1 < 32) && wv == 0)
      tdm_load_wpanel(&wpanel[buf ^ 1][0][0], Wbf + (c + 1) * 32);
    // Double-buffer next A fragment from global.
    if (c + 1 < 32) {
      const unsigned short* ap = aRow + (c + 1) * 32;
      __builtin_prefetch(ap + 32);
      aBuf[buf ^ 1].h[0] = ld8(ap);
      aBuf[buf ^ 1].h[1] = ld8(ap + 8);
    }
#pragma unroll
    for (int nt = 0; nt < 16; ++nt) {
      // B fragment (16-bit B 32x16): lane l15 = column; hlf picks K 0-15/16-31.
      const unsigned short* bp = &wpanel[buf][nt * 16 + l15][hlf * 16];
      ABf b;
      b.h[0] = *(const bf16x8*)bp;
      b.h[1] = *(const bf16x8*)(bp + 8);
      accP[nt] = __builtin_amdgcn_wmma_f32_16x16x32_bf16(
          false, aBuf[buf].v, false, b.v, (short)0, accP[nt], false, false);
    }
    if (wv == 0) __builtin_amdgcn_s_wait_tensorcnt(0);
    __syncthreads();   // next panel landed; buffer `buf` free for reuse
  }

  // ---------------- square + C-layout -> A-layout via LDS --------------------
#pragma unroll
  for (int nt = 0; nt < 16; ++nt) {
#pragma unroll
    for (int i = 0; i < 8; ++i) {
      float p = accP[nt][i];
      qlds[wv][i + hlf * 8][nt * 16 + l15] = f2bf(p * p);
    }
  }
  __syncthreads();

  ABf aq[8];
#pragma unroll
  for (int kk = 0; kk < 8; ++kk) {
    const unsigned short* qp = &qlds[wv][l15][kk * 32 + hlf * 16];
    aq[kk].h[0] = *(const bf16x8*)qp;
    aq[kk].h[1] = *(const bf16x8*)(qp + 8);
  }

  // ---------------- GEMM2: g[16 x 1024] = q @ U^T + Heun epilogue ------------
  // U B-fragments double-buffered in registers: load tile nt2+1 during nt2's WMMAs.
  const float halfdt = 0.5f * DT;
  ABf bU[2][8];
#pragma unroll
  for (int kk = 0; kk < 8; ++kk) {
    const unsigned short* bp = Ubf + (size_t)l15 * RDIM + kk * 32 + hlf * 16;
    bU[0][kk].h[0] = ld8(bp);
    bU[0][kk].h[1] = ld8(bp + 8);
  }

  for (int nt2 = 0; nt2 < 64; ++nt2) {
    const int cur = nt2 & 1;
    if (nt2 + 1 < 64) {
#pragma unroll
      for (int kk = 0; kk < 8; ++kk) {
        const unsigned short* bp =
            Ubf + (size_t)((nt2 + 1) * 16 + l15) * RDIM + kk * 32 + hlf * 16;
        bU[cur ^ 1][kk].h[0] = ld8(bp);
        bU[cur ^ 1][kk].h[1] = ld8(bp + 8);
      }
    }
    v8f acc2 = zero;
#pragma unroll
    for (int kk = 0; kk < 8; ++kk) {
      acc2 = __builtin_amdgcn_wmma_f32_16x16x32_bf16(
          false, aq[kk].v, false, bU[cur][kk].v, (short)0, acc2, false, false);
    }
#pragma unroll
    for (int i = 0; i < 8; ++i) {
      const int    gm  = row0 + i + hlf * 8;
      const int    n   = nt2 * 16 + l15;
      const size_t idx = (size_t)gm * DDIM + n;
      const float  g   = acc2[i];
      const float  dv  = force[idx] - g;
      const float  vk  = vmaster[idx];
      if (PHASE == 0) {
        const float vp = vk + DT * dv;
        dv1buf[idx]  = dv;
        Sacc[idx]   += vk + vp;
        vout_bf[idx] = f2bf(vp);
      } else {
        const float vn = vk + halfdt * (dv1buf[idx] + dv);
        vmaster[idx] = vn;
        vout_bf[idx] = f2bf(vn);
      }
    }
  }
}

// ---------------------------------------------------------------------------
__global__ __launch_bounds__(256)
void prep_kernel(const float* __restrict__ v_in,
                 const float* __restrict__ U,
                 const float* __restrict__ W,
                 unsigned short* __restrict__ Wbf,
                 unsigned short* __restrict__ Ubf,
                 unsigned short* __restrict__ vbf,
                 float* __restrict__ vmaster,
                 float* __restrict__ Sacc)
{
  const size_t i = (size_t)blockIdx.x * blockDim.x + threadIdx.x;
  if (i < (size_t)DDIM * RDIM) {
    Wbf[i] = f2bf(W[i]);
    Ubf[i] = f2bf(U[i]);
  }
  const float vv = v_in[i];
  vbf[i]     = f2bf(vv);
  vmaster[i] = vv;
  Sacc[i]    = 0.0f;
}

__global__ __launch_bounds__(256)
void finalize_kernel(const float* __restrict__ x_in,
                     float* __restrict__ outx)
{
  const size_t i = (size_t)blockIdx.x * blockDim.x + threadIdx.x;
  const float pi     = 3.14159265358979323846f;
  const float two_pi = 6.28318530717958647692f;
  float t = x_in[i] + (0.5f * DT) * outx[i];
  float w = t + pi;
  w -= floorf(w * (1.0f / two_pi)) * two_pi;
  outx[i] = w - pi;
}

// ---------------------------------------------------------------------------
extern "C" void kernel_launch(void* const* d_in, const int* in_sizes, int n_in,
                              void* d_out, int out_size, void* d_ws, size_t ws_size,
                              hipStream_t stream) {
  (void)in_sizes; (void)n_in; (void)out_size; (void)ws_size;

  const float* x     = (const float*)d_in[0];
  const float* v     = (const float*)d_in[1];
  const float* force = (const float*)d_in[2];
  const float* U     = (const float*)d_in[3];   // [1024][256]
  const float* W     = (const float*)d_in[4];   // [256][1024]

  const size_t nBD = (size_t)BDIM * DDIM;
  const size_t nWU = (size_t)DDIM * RDIM;

  unsigned short* Wbf  = (unsigned short*)d_ws;
  unsigned short* Ubf  = Wbf + nWU;
  unsigned short* vbf  = Ubf + nWU;
  unsigned short* vpbf = vbf + nBD;
  float*          dv1  = (float*)(vpbf + nBD);

  float* Sacc    = (float*)d_out;               // S during loop -> wrapped x
  float* vmaster = (float*)d_out + nBD;         // fp32 v state -> final v

  prep_kernel<<<(unsigned)(nBD / 256), 256, 0, stream>>>(
      v, U, W, Wbf, Ubf, vbf, vmaster, Sacc);

  const unsigned grid = BDIM / 32;
  for (int s = 0; s < 16; ++s) {
    christoffel_step<0><<<grid, 64, 0, stream>>>(
        vbf, vmaster, force, Wbf, Ubf, dv1, Sacc, vpbf);
    christoffel_step<1><<<grid, 64, 0, stream>>>(
        vpbf, vmaster, force, Wbf, Ubf, dv1, Sacc, vbf);
  }

  finalize_kernel<<<(unsigned)(nBD / 256), 256, 0, stream>>>(x, Sacc);
}
